// Spiking_14010183319853
// MI455X (gfx1250) — matrix operand order repaired and checked by reading, not compiled, not run on hardware
//
#include <hip/hip_runtime.h>

typedef __attribute__((ext_vector_type(16))) _Float16 v16h;
typedef __attribute__((ext_vector_type(8)))  _Float16 v8h;
typedef __attribute__((ext_vector_type(4)))  _Float16 v4h;
typedef __attribute__((ext_vector_type(8)))  float    v8f;

#define T_STEPS 256
#define B_SZ    128
#define I_SZ    1024
#define O_SZ    512
#define M_TOT   (T_STEPS * B_SZ)   // 32768
#define K_TOT   I_SZ               // 1024
#define BLOCK_M 64                 // rows of X per block (4 WMMA M-tiles)
#define LDS_STRIDE (K_TOT + 8)     // +16B/row: bank step 4 -> conflict-free b128
#define GEMM_THREADS 512           // 16 waves

// ---------------- Kernel 1: weight f32 -> f16 copy in workspace -------------
__global__ void cvt_weight_f16(const float* __restrict__ w,
                               _Float16* __restrict__ wh, int n4) {
  int gid = blockIdx.x * blockDim.x + threadIdx.x;
  if (gid < n4) {
    float4 v = ((const float4*)w)[gid];
    v4h h;
    h[0] = (_Float16)v.x; h[1] = (_Float16)v.y;
    h[2] = (_Float16)v.z; h[3] = (_Float16)v.w;
    ((v4h*)wh)[gid] = h;
  }
}

// ---------------- Kernel 2: Z = X * W^T + bias via v_wmma_f32_16x16x32_f16 --
// Block: 512 threads = 16 waves. Each block: 64 rows of X (staged once to LDS
// as f16 with padded stride; x read from HBM exactly once) x full N=512.
// Waves 0-7 own rows [0,32), waves 8-15 rows [32,64); wave w owns N in
// [64*(w%8), +64): 2 M-subtiles x 4 N-tiles = 8 f32 accumulators per wave.
__launch_bounds__(GEMM_THREADS)
__global__ void snn_gemm_wmma(const float* __restrict__ x,
                              const _Float16* __restrict__ wh,
                              const float* __restrict__ bias,
                              float* __restrict__ z) {
  __shared__ _Float16 As[BLOCK_M * LDS_STRIDE];   // 64*1032*2 = 132,096 B

  const int tid  = threadIdx.x;
  const int wave = tid >> 5;
  const int lane = tid & 31;
  const int lhi  = lane >> 4;           // 0/1: which half-wave
  const int llo  = lane & 15;
  const int wrow = (wave >> 3) * 32;    // row base for this wave
  const int n_wave = (wave & 7) * 64;   // N base for this wave
  const size_t mbase = (size_t)blockIdx.x * BLOCK_M;

  // ---- Stage A tile: 64x1024 f32 -> f16 in LDS (coalesced float4 loads) ----
  const float4* xg = (const float4*)(x + mbase * (size_t)K_TOT);
  #pragma unroll
  for (int it = 0; it < (BLOCK_M * K_TOT / 4) / GEMM_THREADS; ++it) {
    int idx4 = it * GEMM_THREADS + tid;
    float4 v = xg[idx4];
    v4h h;
    h[0] = (_Float16)v.x; h[1] = (_Float16)v.y;
    h[2] = (_Float16)v.z; h[3] = (_Float16)v.w;
    int e   = idx4 * 4;
    int row = e >> 10;          // /1024
    int col = e & (K_TOT - 1);
    *(v4h*)(As + (size_t)row * LDS_STRIDE + col) = h;
  }
  __syncthreads();

  v8f acc[2][4];
  #pragma unroll
  for (int mi = 0; mi < 2; ++mi)
    #pragma unroll
    for (int ni = 0; ni < 4; ++ni)
      acc[mi][ni] = (v8f){0.f,0.f,0.f,0.f,0.f,0.f,0.f,0.f};

  for (int kk = 0; kk < K_TOT; kk += 32) {
    // A fragments: lane holds row (wrow + 16*mi + llo),
    // chunks K=[k0,k0+8) and [k0+16,k0+24), k0 = kk + 8*lhi
    v16h a[2];
    #pragma unroll
    for (int mi = 0; mi < 2; ++mi) {
      const _Float16* ap =
          As + (size_t)(wrow + mi * 16 + llo) * LDS_STRIDE + kk + lhi * 8;
      v8h lo = *(const v8h*)(ap);
      v8h hi = *(const v8h*)(ap + 16);
      a[mi] = __builtin_shufflevector(lo, hi,
                 0,1,2,3,4,5,6,7,8,9,10,11,12,13,14,15);
    }
    // B fragments: lane holds column n, K chunks [k0,k0+8) and [k0+16,k0+24) of row n of W (f16)
    v16h b[4];
    #pragma unroll
    for (int ni = 0; ni < 4; ++ni) {
      int n = n_wave + ni * 16 + llo;
      const _Float16* bp = wh + (size_t)n * K_TOT + kk + lhi * 8;
      v8h blo = *(const v8h*)(bp);
      v8h bhi = *(const v8h*)(bp + 16);
      b[ni] = __builtin_shufflevector(blo, bhi,
                 0,1,2,3,4,5,6,7,8,9,10,11,12,13,14,15);
    }
    #pragma unroll
    for (int mi = 0; mi < 2; ++mi)
      #pragma unroll
      for (int ni = 0; ni < 4; ++ni)
        acc[mi][ni] = __builtin_amdgcn_wmma_f32_16x16x32_f16(
            false, a[mi], false, b[ni], (short)0, acc[mi][ni], false, false);
  }

  // ---- Epilogue: add bias, store Z into d_out's output region --------------
  #pragma unroll
  for (int mi = 0; mi < 2; ++mi) {
    #pragma unroll
    for (int ni = 0; ni < 4; ++ni) {
      int n = n_wave + ni * 16 + llo;
      float bv = bias[n];
      #pragma unroll
      for (int j = 0; j < 8; ++j) {
        int mloc = wrow + mi * 16 + lhi * 8 + j;
        z[(mbase + mloc) * (size_t)O_SZ + n] = acc[mi][ni][j] + bv;
      }
    }
  }
}

// ---------------- Kernel 3: sequential LIF scan (element-wise) --------------
// One thread per (b, o); reads Z in place and overwrites with spikes.
__global__ void snn_scan(const float* __restrict__ wl,
                         float* __restrict__ out,      // [T,B,O], Z -> spikes
                         float* __restrict__ tpost) {  // [B,O]
  int gid = blockIdx.x * blockDim.x + threadIdx.x;     // 0..B*O-1
  int o = gid & (O_SZ - 1);
  const float w = wl[o];
  float ep = 0.f, u = 0.f, sp = 0.f, vref = 0.f, tp = -1.f;
  float* p = out + (size_t)gid;
  const size_t stride = (size_t)B_SZ * O_SZ;
  for (int t = 0; t < T_STEPS; ++t) {
    float zv = __builtin_nontemporal_load(p + (size_t)t * stride);
    ep = ep * 0.8f + (zv + sp * w);          // decay_s = 1 - 1/5
    u  = u * 0.9f + ep * 0.2f;               // decay_m = 1 - 1/10; ep/TAU_S
    u  = (vref > 0.f) ? 0.f : u;             // refractory gate (old vref)
    sp = (u > 1.f) ? 1.f : 0.f;              // spike
    vref = fminf(fmaxf(2.f * sp + (vref - 1.f), 0.f), 2.f);
    tp = fmaxf(sp * (float)(t + 1) - 1.f, tp);
    __builtin_nontemporal_store(sp, p + (size_t)t * stride);
  }
  __builtin_nontemporal_store(tp, tpost + gid);
}

// ---------------- Kernel 4: times_pre max-scan over x -----------------------
__global__ void snn_tpre(const float* __restrict__ x,
                         float* __restrict__ tpre) {   // [B,I]
  int gid = blockIdx.x * blockDim.x + threadIdx.x;     // 0..B*I-1
  const float* p = x + (size_t)gid;
  const size_t stride = (size_t)B_SZ * I_SZ;
  float tp = -1.f;
  for (int t = 0; t < T_STEPS; ++t) {
    float xv = __builtin_nontemporal_load(p + (size_t)t * stride);
    tp = fmaxf(xv * (float)(t + 1) - 1.f, tp);
  }
  __builtin_nontemporal_store(tp, tpre + gid);
}

extern "C" void kernel_launch(void* const* d_in, const int* in_sizes, int n_in,
                              void* d_out, int out_size, void* d_ws, size_t ws_size,
                              hipStream_t stream) {
  const float* x  = (const float*)d_in[0];   // [T,B,I]
  const float* w  = (const float*)d_in[1];   // [O,I]
  const float* bi = (const float*)d_in[2];   // [O]
  const float* wl = (const float*)d_in[3];   // [O]

  float* out   = (float*)d_out;
  float* zbuf  = out;                                        // [T,B,O] (in-place)
  float* tpre  = out + (size_t)T_STEPS * B_SZ * O_SZ;        // [B,I]
  float* tpost = tpre + (size_t)B_SZ * I_SZ;                 // [B,O]

  _Float16* wh = (_Float16*)d_ws;                            // 1 MB f16 weights

  cvt_weight_f16<<<(O_SZ * I_SZ / 4 + 255) / 256, 256, 0, stream>>>(w, wh, O_SZ * I_SZ / 4);
  snn_gemm_wmma<<<M_TOT / BLOCK_M, GEMM_THREADS, 0, stream>>>(x, wh, bi, zbuf);
  snn_scan<<<(B_SZ * O_SZ) / 256, 256, 0, stream>>>(wl, zbuf, tpost);
  snn_tpre<<<(B_SZ * I_SZ) / 256, 256, 0, stream>>>(x, tpre);
}
